// AttendAndSpell_68539088109606
// MI455X (gfx1250) — matrix operand order repaired and checked by reading, not compiled, not run on hardware
//
#include <hip/hip_runtime.h>
#include <stdint.h>

// ---------------- types / helpers ----------------
typedef __attribute__((ext_vector_type(16))) __bf16 v16bf;
typedef __attribute__((ext_vector_type(8)))  float  v8f;
typedef unsigned short u16;
typedef unsigned int   u32;

#define DEVINL static __device__ __forceinline__

DEVINL u16 f2bf(float f){
  u32 u = __builtin_bit_cast(u32, f);
  u += 0x7FFFu + ((u >> 16) & 1u);          // round-to-nearest-even
  return (u16)(u >> 16);
}
DEVINL float bf2f(u16 b){
  u32 u = ((u32)b) << 16;
  return __builtin_bit_cast(float, u);
}
DEVINL float sigm(float x){ return 1.0f / (1.0f + __expf(-x)); }

// ---------------- WMMA fragment loaders ----------------
// A 16x32 bf16 fragment (ISA 7.12.2): lane L (and L+16) hold row L%16;
// element i -> k = (i/8)*16 + (lane/16)*8 + (i%8)  => two contiguous 16B loads.
DEVINL v16bf load_a_bf16(const u16* A, long row, int lda, int kk, int half){
  const u16* p = A + row*(long)lda + kk + half*8;
  v16bf a;
  ((uint4*)&a)[0] = *(const uint4*)(p);
  ((uint4*)&a)[1] = *(const uint4*)(p + 16);
  return a;
}
DEVINL v16bf load_a_f32(const float* A, long row, int lda, int kk, int half){
  const float* p = A + row*(long)lda + kk + half*8;
  u16 tmp[16];
#pragma unroll
  for (int i = 0; i < 8; ++i) tmp[i]     = f2bf(p[i]);
#pragma unroll
  for (int i = 0; i < 8; ++i) tmp[8 + i] = f2bf(p[16 + i]);
  v16bf a;
  __builtin_memcpy(&a, tmp, 32);
  return a;
}
// B 32x16 bf16 fragment: lane n%16 holds column n, same k order as A:
// element i -> k = (i/8)*16 + (lane/16)*8 + (i%8) of weight row n (row-major W[n,k] == B^T).
DEVINL v16bf load_b(const u16* W, long n, int ldw, int kk, int half){
  const u16* p = W + n*(long)ldw + kk + half*8;
  v16bf b;
  ((uint4*)&b)[0] = *(const uint4*)(p);
  ((uint4*)&b)[1] = *(const uint4*)(p + 16);
  return b;
}

DEVINL void mma_pair(v8f acc[4], const void* A, int lda, int aF32,
                     const u16* W, int ldw, int K,
                     long row, int colBase, int half, int l16){
  for (int kk = 0; kk < K; kk += 32){
    __builtin_prefetch((const void*)(W + (long)(colBase + l16)*ldw + kk + 32), 0, 3);
    v16bf af;
    if (aF32) af = load_a_f32((const float*)A, row, lda, kk, half);
    else      af = load_a_bf16((const u16*)A, row, lda, kk, half);
#pragma unroll
    for (int t = 0; t < 4; ++t){
      v16bf bf = load_b(W, colBase + t*16 + l16, ldw, kk, half);
      acc[t] = __builtin_amdgcn_wmma_f32_16x16x32_bf16(
                 false, af, false, bf, (short)0, acc[t], false, false);
    }
  }
}

// ---------------- generic fused GEMM ----------------
// out(M,N) = sum_p A_p(M,K_p) @ W_p(N,K_p)^T + bias1 + bias2, optional ReLU.
// Block = 128 threads (4 waves), tile = 64x64, wave = 16x64 strip (4 WMMA accs).
__global__ __launch_bounds__(128)
void gemm_kernel(const void* A1,int lda1,int a1f,const u16* W1,int ldw1,int K1,
                 const void* A2,int lda2,int a2f,const u16* W2,int ldw2,int K2,
                 const void* A3,int lda3,int a3f,const u16* W3,int ldw3,int K3,
                 const float* bias1,const float* bias2,
                 int N, float* outF, long ldoF, u16* outB, long ldoB, int act)
{
  const int nb   = N >> 6;
  const int brow = blockIdx.x / nb;
  const int bcol = blockIdx.x - brow*nb;
  const int wave = threadIdx.x >> 5;
  const int lane = threadIdx.x & 31;
  const int half = lane >> 4;
  const int l16  = lane & 15;
  const long row = (long)brow*64 + wave*16 + l16;
  const int  colBase = bcol*64;

  v8f acc[4];
#pragma unroll
  for (int t = 0; t < 4; ++t){
    v8f z = {0.f,0.f,0.f,0.f,0.f,0.f,0.f,0.f};
    acc[t] = z;
  }

  if (K1) mma_pair(acc, A1, lda1, a1f, W1, ldw1, K1, row, colBase, half, l16);
  if (K2) mma_pair(acc, A2, lda2, a2f, W2, ldw2, K2, row, colBase, half, l16);
  if (K3) mma_pair(acc, A3, lda3, a3f, W3, ldw3, K3, row, colBase, half, l16);

#pragma unroll
  for (int t = 0; t < 4; ++t){
    const int n = colBase + t*16 + l16;
    float bv = 0.f;
    if (bias1) bv += bias1[n];
    if (bias2) bv += bias2[n];
#pragma unroll
    for (int r = 0; r < 8; ++r){
      const long m = (long)brow*64 + wave*16 + half*8 + r;   // C/D layout
      float v = acc[t][r] + bv;
      if (act) v = fmaxf(v, 0.f);
      if (outF) outF[m*ldoF + n] = v;
      if (outB) outB[m*ldoB + n] = f2bf(v);
    }
  }
}

// ---------------- LSTM pointwise ----------------
__global__ void lstm_pointwise(const float* __restrict__ gates,
                               const float* __restrict__ c_in,
                               float* __restrict__ c_out,
                               u16* __restrict__ h_out)
{
  int idx = blockIdx.x*blockDim.x + threadIdx.x;   // B*H = 32768
  int b = idx >> 9;
  int j = idx & 511;
  const float* g = gates + (long)b*2048;
  float iv = sigm(g[j]);
  float fv = sigm(g[512 + j]);
  float gv = tanhf(g[1024 + j]);
  float ov = sigm(g[1536 + j]);
  float c  = fv * c_in[idx] + iv * gv;
  c_out[idx] = c;
  h_out[idx] = f2bf(ov * tanhf(c));
}

// ---------------- attention (per-batch block) ----------------
__global__ __launch_bounds__(256)
void attention_kernel(const float* __restrict__ seF,
                      const u16* __restrict__ heB,
                      u16* __restrict__ ctxB)
{
  __shared__ float eL[1024];
  __shared__ float red[256];
  const int b = blockIdx.x;
  const int tid = threadIdx.x;
  const int wv = tid >> 5;
  const int lane = tid & 31;

  float sreg[16];
  const float* sp = seF + (long)b*512 + lane*16;
#pragma unroll
  for (int i = 0; i < 16; ++i) sreg[i] = sp[i];

  const u16* hb = heB + (long)b*1024*512;
  for (int l = wv; l < 1024; l += 8){
    const u16* hp = hb + (long)l*512 + lane*16;
    uint4 u0 = ((const uint4*)hp)[0];
    uint4 u1 = ((const uint4*)hp)[1];
    u32 w[8] = {u0.x,u0.y,u0.z,u0.w,u1.x,u1.y,u1.z,u1.w};
    float d = 0.f;
#pragma unroll
    for (int i = 0; i < 8; ++i){
      d += bf2f((u16)(w[i] & 0xffffu)) * sreg[2*i];
      d += bf2f((u16)(w[i] >> 16))     * sreg[2*i + 1];
    }
#pragma unroll
    for (int off = 16; off; off >>= 1) d += __shfl_xor(d, off, 32);
    if (lane == 0) eL[l] = d;
  }
  __syncthreads();

  float m = -3.0e38f;
  for (int l = tid; l < 1024; l += 256) m = fmaxf(m, eL[l]);
  red[tid] = m; __syncthreads();
  for (int s2 = 128; s2; s2 >>= 1){
    if (tid < s2) red[tid] = fmaxf(red[tid], red[tid + s2]);
    __syncthreads();
  }
  m = red[0]; __syncthreads();

  float s = 0.f;
  for (int l = tid; l < 1024; l += 256){
    float ev = __expf(eL[l] - m);
    eL[l] = ev;
    s += ev;
  }
  red[tid] = s; __syncthreads();
  for (int s2 = 128; s2; s2 >>= 1){
    if (tid < s2) red[tid] += red[tid + s2];
    __syncthreads();
  }
  float inv = 1.f / red[0];
  __syncthreads();
  for (int l = tid; l < 1024; l += 256) eL[l] *= inv;
  __syncthreads();

  for (int h = tid; h < 512; h += 256){
    float acc = 0.f;
    for (int l = 0; l < 1024; ++l)
      acc += eL[l] * bf2f(hb[(long)l*512 + h]);
    ctxB[(long)b*512 + h] = f2bf(acc);
  }
}

// ---------------- small utility kernels ----------------
__global__ void cvt_kernel(const float* __restrict__ in, u16* __restrict__ out, long n){
  long i = (long)blockIdx.x*256 + threadIdx.x;
  if (i < n) out[i] = f2bf(in[i]);
}
__global__ void zero_kernel(u32* __restrict__ p, long n){
  long i = (long)blockIdx.x*256 + threadIdx.x;
  if (i < n) p[i] = 0u;
}
__global__ void embed_kernel(const int* __restrict__ y, const float* __restrict__ emb,
                             u16* __restrict__ out){
  long idx = (long)blockIdx.x*256 + threadIdx.x;   // B*S*E
  int e  = (int)(idx & 255);
  long bt = idx >> 8;
  int tok = y[bt];
  out[idx] = f2bf(emb[(long)tok*256 + e]);
}

// ---------------- host-side launch helper ----------------
static inline void gemm_launch(hipStream_t s,
    const void* A1,int lda1,int a1f,const u16* W1,int ldw1,int K1,
    const void* A2,int lda2,int a2f,const u16* W2,int ldw2,int K2,
    const void* A3,int lda3,int a3f,const u16* W3,int ldw3,int K3,
    const float* b1,const float* b2,int M,int N,
    float* outF,long ldoF,u16* outB,long ldoB,int act)
{
  dim3 grid((unsigned)((M/64)*(N/64)));
  gemm_kernel<<<grid, 128, 0, s>>>(A1,lda1,a1f,W1,ldw1,K1,
                                   A2,lda2,a2f,W2,ldw2,K2,
                                   A3,lda3,a3f,W3,ldw3,K3,
                                   b1,b2,N,outF,ldoF,outB,ldoB,act);
}

extern "C" void kernel_launch(void* const* d_in, const int* in_sizes, int n_in,
                              void* d_out, int out_size, void* d_ws, size_t ws_size,
                              hipStream_t stream)
{
  (void)in_sizes; (void)n_in; (void)out_size; (void)ws_size;
  const int B = 64, S = 128, E = 256, H = 512, L = 1024, V = 64;
  const int G4 = 4 * H;  // 2048

  const int*   y    = (const int*)  d_in[0];
  const float* enc  = (const float*)d_in[1];
  const float* emb  = (const float*)d_in[2];
  const float* aWih = (const float*)d_in[3];
  const float* aWhh = (const float*)d_in[4];
  const float* abih = (const float*)d_in[5];
  const float* abhh = (const float*)d_in[6];
  const float* rWih = (const float*)d_in[7];
  const float* rWhh = (const float*)d_in[8];
  const float* rbih = (const float*)d_in[9];
  const float* rbhh = (const float*)d_in[10];
  const float* Wsw  = (const float*)d_in[11];
  const float* Wsb  = (const float*)d_in[12];
  const float* Whw  = (const float*)d_in[13];
  const float* Whb  = (const float*)d_in[14];
  const float* W1   = (const float*)d_in[15];
  const float* b1   = (const float*)d_in[16];
  const float* W2   = (const float*)d_in[17];
  const float* b2   = (const float*)d_in[18];
  float* out = (float*)d_out;

  // -------- workspace layout (~89 MB) --------
  char* ws = (char*)d_ws;
  size_t off = 0;
  auto take = [&](size_t bytes) -> char* {
    char* p = ws + off;
    off += (bytes + 255) & ~(size_t)255;
    return p;
  };
  u16* heB    = (u16*)take((size_t)B*L*H*2);       // h_e bf16 (L2-resident, 64MB)
  u16* yembB  = (u16*)take((size_t)B*S*E*2);
  u16* aWihB  = (u16*)take((size_t)G4*(E+H)*2);
  u16* aWhhB  = (u16*)take((size_t)G4*H*2);
  u16* rWihB  = (u16*)take((size_t)2*G4*H*2);
  u16* rWhhB  = (u16*)take((size_t)2*G4*H*2);
  u16* WswB   = (u16*)take((size_t)H*H*2);
  u16* WhwB   = (u16*)take((size_t)H*H*2);
  u16* W1B    = (u16*)take((size_t)H*2*H*2);
  u16* W2B    = (u16*)take((size_t)V*H*2);
  float* c0   = (float*)take((size_t)B*H*4);       // states start (contiguous)
  float* c1   = (float*)take((size_t)B*H*4);
  float* c2   = (float*)take((size_t)B*H*4);
  u16* h0B    = (u16*)take((size_t)B*H*2);
  u16* h1B    = (u16*)take((size_t)B*H*2);
  u16* h2B    = (u16*)take((size_t)B*H*2);
  u16* ctxB   = (u16*)take((size_t)B*H*2);
  float* gatesF = (float*)take((size_t)B*G4*4);
  float* seF    = (float*)take((size_t)B*H*4);
  u16* hidB   = (u16*)take((size_t)B*H*2);

  // -------- one-time prep --------
  auto cvt = [&](const float* src, u16* dst, long n){
    cvt_kernel<<<(unsigned)((n + 255) / 256), 256, 0, stream>>>(src, dst, n);
  };
  cvt(aWih, aWihB, (long)G4*(E+H));
  cvt(aWhh, aWhhB, (long)G4*H);
  cvt(rWih, rWihB, 2L*G4*H);
  cvt(rWhh, rWhhB, 2L*G4*H);
  cvt(Wsw,  WswB,  (long)H*H);
  cvt(Whw,  WhwB,  (long)H*H);
  cvt(W1,   W1B,   (long)H*2*H);
  cvt(W2,   W2B,   (long)V*H);

  embed_kernel<<<(B*S*E)/256, 256, 0, stream>>>(y, emb, yembB);

  { // zero all recurrent state (c0..ctxB contiguous, sizes are 256B multiples)
    long words = ((long)B*H*4*3 + (long)B*H*2*4) / 4;
    zero_kernel<<<(unsigned)((words + 255)/256), 256, 0, stream>>>((u32*)c0, words);
  }

  // h_e = encoder_h @ Wh_w^T + Wh_b   (hoisted; fp32 A converted in-register)
  gemm_launch(stream, enc, H, 1, WhwB, H, H,
              nullptr,0,0,nullptr,0,0, nullptr,0,0,nullptr,0,0,
              Whb, nullptr, B*L, H, nullptr, 0, heB, H, 0);

  // -------- sequential decode (stream-ordered dependencies) --------
  for (int t = 0; t < S; ++t){
    // attention LSTM cell: gates = [x_t | ctx] @ Wih^T + h0 @ Whh^T + bih + bhh
    gemm_launch(stream,
      yembB + (size_t)t*E, S*E, 0, aWihB,     E+H, E,
      ctxB,               H,   0, aWihB + E,  E+H, H,
      h0B,                H,   0, aWhhB,      H,   H,
      abih, abhh, B, G4, gatesF, G4, nullptr, 0, 0);
    lstm_pointwise<<<(B*H)/256, 256, 0, stream>>>(gatesF, c0, c0, h0B);

    // rnn layer 0
    gemm_launch(stream,
      h0B, H, 0, rWihB, H, H,
      h1B, H, 0, rWhhB, H, H,
      nullptr,0,0,nullptr,0,0,
      rbih, rbhh, B, G4, gatesF, G4, nullptr, 0, 0);
    lstm_pointwise<<<(B*H)/256, 256, 0, stream>>>(gatesF, c1, c1, h1B);

    // rnn layer 1
    gemm_launch(stream,
      h1B, H, 0, rWihB + (size_t)G4*H, H, H,
      h2B, H, 0, rWhhB + (size_t)G4*H, H, H,
      nullptr,0,0,nullptr,0,0,
      rbih + G4, rbhh + G4, B, G4, gatesF, G4, nullptr, 0, 0);
    lstm_pointwise<<<(B*H)/256, 256, 0, stream>>>(gatesF, c2, c2, h2B);

    // s_e = h2 @ Ws_w^T + Ws_b
    gemm_launch(stream,
      h2B, H, 0, WswB, H, H,
      nullptr,0,0,nullptr,0,0, nullptr,0,0,nullptr,0,0,
      Wsb, nullptr, B, H, seF, H, nullptr, 0, 0);

    // softmax attention -> ctx (bf16)
    attention_kernel<<<B, 256, 0, stream>>>(seF, heB, ctxB);

    // hidden = relu([h2 | ctx] @ W1^T + b1)
    gemm_launch(stream,
      h2B,  H, 0, W1B,     2*H, H,
      ctxB, H, 0, W1B + H, 2*H, H,
      nullptr,0,0,nullptr,0,0,
      b1, nullptr, B, H, nullptr, 0, hidB, H, 1);

    // logits -> d_out[(b*S + t)*V + v]
    gemm_launch(stream,
      hidB, H, 0, W2B, H, H,
      nullptr,0,0,nullptr,0,0, nullptr,0,0,nullptr,0,0,
      b2, nullptr, B, V, out + (size_t)t*V, (long)S*V, nullptr, 0, 0);
  }
}
